// Network_4D_8100308320795
// MI455X (gfx1250) — matrix operand order repaired and checked by reading, not compiled, not run on hardware
//
#include <hip/hip_runtime.h>
#include <hip/hip_bf16.h>

// ---------------------------------------------------------------------------
// CDNA5 (gfx1250) dense 4D U-Net as implicit-GEMM BF16 WMMA.
// wave32; v_wmma_f32_16x16x32_bf16; TDM tensor_load_to_lds staging (verified
// to lower in round 1) + s_wait_tensorcnt; weights repacked to [tap][M][Kp]
// (Kp = roundup32) so A-fragments are two b128 loads, no per-element guards.
// ---------------------------------------------------------------------------

typedef __attribute__((ext_vector_type(16))) __bf16 v16bf;
typedef __attribute__((ext_vector_type(8)))  __bf16 v8bf;
typedef __attribute__((ext_vector_type(16))) unsigned short v16us;
typedef __attribute__((ext_vector_type(8)))  float  v8f;
typedef __attribute__((ext_vector_type(4)))  unsigned int v4u;
typedef __attribute__((ext_vector_type(8)))  int    v8i_;
typedef __attribute__((ext_vector_type(4)))  int    v4i_;

#if __has_builtin(__builtin_amdgcn_tensor_load_to_lds) && __has_builtin(__builtin_amdgcn_s_wait_tensorcnt)
#define HAVE_TDM 1
#endif

__device__ __forceinline__ float sigf(float x) { return 1.0f / (1.0f + __expf(-x)); }

// Build a 16x32 bf16 A-fragment from a [*, Kp] row:
// elements j=0..15 are K = kg*16 + j (contiguous).
__device__ __forceinline__ v16bf load_a_frag(const __bf16* __restrict__ row, int kg) {
  const v8bf lo = *(const v8bf*)(row + kg * 16);       // 16B aligned (Kp mult of 32)
  const v8bf hi = *(const v8bf*)(row + kg * 16 + 8);
  return __builtin_shufflevector(lo, hi, 0, 1, 2, 3, 4, 5, 6, 7,
                                 8, 9, 10, 11, 12, 13, 14, 15);
}

// ---------------------------------------------------------------------------
// Conversion / repack kernels (run every call; no caching)
// ---------------------------------------------------------------------------
__global__ __launch_bounds__(256) void cvt_bf16_kernel(const float* __restrict__ x,
                                                       __bf16* __restrict__ o, long n) {
  long i = (long)blockIdx.x * 256 + threadIdx.x;
  if (i < n) o[i] = (__bf16)x[i];
}

// in:  f32 [M][K][taps]   (reference layout, trailing dims collapsed)
// out: bf16 [taps][M][Kp] zero-padded to Kp = roundup32(K)
__global__ __launch_bounds__(256) void repack_weight_kernel(const float* __restrict__ w,
                                                            __bf16* __restrict__ o,
                                                            int M, int K, int Kp, int taps) {
  long i = (long)blockIdx.x * 256 + threadIdx.x;
  long tot = (long)taps * M * Kp;
  if (i >= tot) return;
  int kp = (int)(i % Kp); long q = i / Kp;
  int m = (int)(q % M);
  int tap = (int)(q / M);
  float v = (kp < K) ? w[((size_t)m * K + kp) * taps + tap] : 0.0f;
  o[i] = (__bf16)v;
}

__global__ __launch_bounds__(256) void zero_kernel(float* p, int n) {
  int i = blockIdx.x * 256 + threadIdx.x;
  if (i < n) p[i] = 0.0f;
}

// ---------------------------------------------------------------------------
// TDM 2-D tile load (global [K,N] slice -> LDS), D# per ISA ch.8.
// OOB rows/cols are zero-filled by the TDM.
// ---------------------------------------------------------------------------
#ifdef HAVE_TDM
__device__ __forceinline__ void tdm_load_tile(unsigned lds_addr, unsigned long long gaddr,
                                              unsigned tensor_d0, unsigned tensor_d1,
                                              unsigned tile_d0, unsigned tile_d1,
                                              unsigned long long stride0_elems) {
  v4u g0;
  g0[0] = 1u;                                   // count=1 (valid user descriptor)
  g0[1] = lds_addr;                             // lds_addr [63:32]
  g0[2] = (unsigned)(gaddr & 0xffffffffu);      // global_addr lo
  g0[3] = (unsigned)((gaddr >> 32) & 0x01ffffffu) | (2u << 30);  // ga hi | type=2
  v8i_ g1;
  g1[0] = (int)(1u << 16);                      // workgroup_mask=0, data_size=1 (2B)
  g1[1] = (int)((tensor_d0 & 0xffffu) << 16);   // tensor_dim0[15:0] at [63:48]
  g1[2] = (int)(((tensor_d0 >> 16) & 0xffffu) | ((tensor_d1 & 0xffffu) << 16));
  g1[3] = (int)(((tensor_d1 >> 16) & 0xffffu) | ((tile_d0 & 0xffffu) << 16));
  g1[4] = (int)(tile_d1 & 0xffffu);             // tile_dim1; tile_dim2=0
  g1[5] = (int)(stride0_elems & 0xffffffffu);   // tensor_dim0_stride lo
  g1[6] = (int)((stride0_elems >> 32) & 0xffffu);
  g1[7] = 0;
  v4i_ z4 = {0, 0, 0, 0};
#if defined(__clang_major__) && __clang_major__ >= 23
  v8i_ z8 = {0, 0, 0, 0, 0, 0, 0, 0};
  __builtin_amdgcn_tensor_load_to_lds(g0, g1, z4, z4, z8, 0);
#else
  __builtin_amdgcn_tensor_load_to_lds(g0, g1, z4, z4, 0);
#endif
}
#endif

// ---------------------------------------------------------------------------
// 1x1 conv as GEMM: Out[M,N] = W[M,Kp] @ Act[K,N], bf16 in, f32 out.
// 4 waves/block (64 output cols); 64-deep K tile staged per TDM transfer,
// two WMMAs per barrier. B-fragments come from LDS (rows pre-zeroed by TDM).
// ---------------------------------------------------------------------------
__global__ __launch_bounds__(128) void wmma_gemm_kernel(const __bf16* __restrict__ Wt,
                                                        const __bf16* __restrict__ act,
                                                        float* __restrict__ out,
                                                        int M, int N, int K) {
  __shared__ __bf16 smem[64 * 64];
  const int Kp    = (K + 31) & ~31;
  const int tid   = threadIdx.x;
  const int wave  = tid >> 5;
  const int lane  = tid & 31;
  const int col16 = lane & 15;
  const int kg    = lane >> 4;
  const int nb    = blockIdx.x * 64;
  const int m0    = blockIdx.y * 16;
  const int mrow  = m0 + col16;
  const int col   = wave * 16 + col16;

  v8f c = {0.f, 0.f, 0.f, 0.f, 0.f, 0.f, 0.f, 0.f};

  for (int k0 = 0; k0 < Kp; k0 += 64) {
    __syncthreads();  // previous tile fully consumed
#ifdef HAVE_TDM
    if (wave == 0) {
      unsigned long long ga =
          (unsigned long long)(const void*)act + ((unsigned long long)(size_t)k0 * N + nb) * 2ull;
      tdm_load_tile((unsigned)(size_t)(void*)smem, ga,
                    (unsigned)(N - nb), (unsigned)(K - k0), 64u, 64u,
                    (unsigned long long)N);
      __builtin_amdgcn_s_wait_tensorcnt(0);
    }
#else
    for (int idx = tid; idx < 64 * 64; idx += 128) {
      int kk = idx >> 6, nn = idx & 63;
      __bf16 v = (__bf16)0.f;
      if ((k0 + kk) < K && (nb + nn) < N) v = act[(size_t)(k0 + kk) * N + nb + nn];
      smem[idx] = v;
    }
#endif
    if (wave == 1 && (k0 + 64) < K)
      __builtin_prefetch(act + (size_t)(k0 + 64) * N + nb, 0, 3);  // global_prefetch_b8
    __syncthreads();

    // ---- chunk 0 (K rows k0 .. k0+31) ----
    {
      v16bf a = load_a_frag(Wt + (size_t)mrow * Kp + k0, kg);
      v16us bu;
#pragma unroll
      for (int j = 0; j < 16; ++j)
        bu[j] = *(const unsigned short*)&smem[(kg * 16 + j) * 64 + col];
      v16bf b = __builtin_bit_cast(v16bf, bu);
      c = __builtin_amdgcn_wmma_f32_16x16x32_bf16(false, a, false, b, (short)0, c, false, false);
    }
    // ---- chunk 1 (K rows k0+32 .. k0+63), uniform branch ----
    if (k0 + 32 < Kp) {
      v16bf a = load_a_frag(Wt + (size_t)mrow * Kp + k0 + 32, kg);
      v16us bu;
#pragma unroll
      for (int j = 0; j < 16; ++j)
        bu[j] = *(const unsigned short*)&smem[(32 + kg * 16 + j) * 64 + col];
      v16bf b = __builtin_bit_cast(v16bf, bu);
      c = __builtin_amdgcn_wmma_f32_16x16x32_bf16(false, a, false, b, (short)0, c, false, false);
    }
  }

  const int ncol = nb + col;
  if (ncol < N) {
#pragma unroll
    for (int r = 0; r < 8; ++r)                  // D: VGPR r -> M = r + 8*(lane>=16)
      out[(size_t)(m0 + r + 8 * kg) * N + ncol] = c[r];
  }
}

// ---------------------------------------------------------------------------
// Implicit-GEMM conv (3x3x3 spatial, taps=27 / 1x1x1x3 temporal w/ dilation).
// One wave -> 16(M) x 32(N): two D tiles share one A fragment (A traffic /2,
// 2 WMMAs per K-chunk per tap). Weights: [tap][M][Kp], zero-padded.
// B gathered from global with clamped addresses + selects (no exec branches).
// ---------------------------------------------------------------------------
__global__ __launch_bounds__(32) void wmma_conv_kernel(const __bf16* __restrict__ Wt,
                                                       const __bf16* __restrict__ act,
                                                       float* __restrict__ out,
                                                       int M, int K,
                                                       int Xd, int Yd, int Zd, int Td,
                                                       int ntaps, int mode, int dil) {
  const int lane  = threadIdx.x & 31;
  const int col16 = lane & 15;
  const int kg    = lane >> 4;
  const int Kp    = (K + 31) & ~31;
  const long N    = (long)Xd * Yd * Zd * Td;
  const long n0   = (long)blockIdx.x * 32;
  const int  m0   = blockIdx.y * 16;
  const int  mrow = m0 + col16;

  // decode the two output voxels this lane covers
  int cx[2], cy[2], cz[2], ct[2];
  long ncs[2] = {n0 + col16, n0 + 16 + col16};
  bool live[2];
#pragma unroll
  for (int u = 0; u < 2; ++u) {
    live[u] = ncs[u] < N;
    long nv = live[u] ? ncs[u] : 0;
    ct[u] = (int)(nv % Td); long q = nv / Td;
    cz[u] = (int)(q % Zd); q /= Zd;
    cy[u] = (int)(q % Yd);
    cx[u] = (int)(q / Yd);
  }

  v8f c0 = {0.f, 0.f, 0.f, 0.f, 0.f, 0.f, 0.f, 0.f};
  v8f c1 = {0.f, 0.f, 0.f, 0.f, 0.f, 0.f, 0.f, 0.f};

  for (int tap = 0; tap < ntaps; ++tap) {
    int dx = 0, dy = 0, dz = 0, dt = 0;
    if (mode == 0) { dx = tap / 9 - 1; dy = (tap / 3) % 3 - 1; dz = tap % 3 - 1; }
    else           { dt = (tap - 1) * dil; }

    bool v[2]; long src[2];
#pragma unroll
    for (int u = 0; u < 2; ++u) {
      int xs = cx[u] + dx, ys = cy[u] + dy, zs = cz[u] + dz, ts = ct[u] + dt;
      v[u] = live[u] && xs >= 0 && xs < Xd && ys >= 0 && ys < Yd &&
             zs >= 0 && zs < Zd && ts >= 0 && ts < Td;
      long s = (((long)xs * Yd + ys) * Zd + zs) * Td + ts;
      src[u] = v[u] ? s : 0;                     // clamp: always a safe address
    }
    __builtin_prefetch(act + src[0], 0, 3);

    const __bf16* wrow = Wt + ((size_t)tap * M + mrow) * Kp;
    for (int k0 = 0; k0 < Kp; k0 += 32) {
      v16bf a = load_a_frag(wrow + k0, kg);      // unguarded: weights zero-padded
      v16us b0u, b1u;
#pragma unroll
      for (int j = 0; j < 16; ++j) {
        int kk = k0 + kg * 16 + j;
        int kc = (kk < K) ? kk : 0;              // clamped row, unpredicated loads
        unsigned short t0 = *(const unsigned short*)(act + (size_t)kc * N + src[0]);
        unsigned short t1 = *(const unsigned short*)(act + (size_t)kc * N + src[1]);
        b0u[j] = (v[0] && kk < K) ? t0 : (unsigned short)0;
        b1u[j] = (v[1] && kk < K) ? t1 : (unsigned short)0;
      }
      v16bf b0 = __builtin_bit_cast(v16bf, b0u);
      v16bf b1 = __builtin_bit_cast(v16bf, b1u);
      c0 = __builtin_amdgcn_wmma_f32_16x16x32_bf16(false, a, false, b0, (short)0, c0, false, false);
      c1 = __builtin_amdgcn_wmma_f32_16x16x32_bf16(false, a, false, b1, (short)0, c1, false, false);
    }
  }

  if (ncs[0] < N) {
#pragma unroll
    for (int r = 0; r < 8; ++r)
      out[(size_t)(m0 + r + 8 * kg) * N + ncs[0]] = c0[r];
  }
  if (ncs[1] < N) {
#pragma unroll
    for (int r = 0; r < 8; ++r)
      out[(size_t)(m0 + r + 8 * kg) * N + ncs[1]] = c1[r];
  }
}

// ---------------------------------------------------------------------------
// BatchNorm: per-channel sum/sumsq via LDS tree + global_atomic_add_f32,
// then fused normalize + leaky-relu emitting f32 and/or bf16.
// ---------------------------------------------------------------------------
__global__ __launch_bounds__(256) void bn_stats_kernel(const float* __restrict__ xf,
                                                       float* __restrict__ mv, int C, long N) {
  __shared__ float ss[256], ss2[256];
  const int c = blockIdx.y;
  float s = 0.f, s2 = 0.f;
  for (long i = (long)blockIdx.x * 256 + threadIdx.x; i < N; i += (long)gridDim.x * 256) {
    float v = xf[(size_t)c * N + i];
    s += v; s2 += v * v;
  }
  ss[threadIdx.x] = s; ss2[threadIdx.x] = s2;
  __syncthreads();
  for (int off = 128; off > 0; off >>= 1) {
    if (threadIdx.x < off) { ss[threadIdx.x] += ss[threadIdx.x + off];
                             ss2[threadIdx.x] += ss2[threadIdx.x + off]; }
    __syncthreads();
  }
  if (threadIdx.x == 0) { atomicAdd(&mv[c], ss[0]); atomicAdd(&mv[C + c], ss2[0]); }
}

__global__ __launch_bounds__(256) void bn_act_kernel(const float* __restrict__ xf,
                                                     const float* __restrict__ mv,
                                                     float* __restrict__ of,
                                                     __bf16* __restrict__ ob,
                                                     int C, long N) {
  long i = (long)blockIdx.x * 256 + threadIdx.x;
  if (i >= (long)C * N) return;
  int c = (int)(i / N);
  float m   = mv[c] / (float)N;
  float var = mv[C + c] / (float)N - m * m;
  float v = (xf[i] - m) * rsqrtf(var + 1e-5f);
  v = (v > 0.f) ? v : 0.01f * v;   // leaky relu 0.01
  if (of) of[i] = v;
  if (ob) ob[i] = (__bf16)v;
}

// ---------------------------------------------------------------------------
// Fused elementwise kernels (bandwidth-bound; single pass, dual emit)
// ---------------------------------------------------------------------------
__global__ __launch_bounds__(256) void blend_kernel(const float* __restrict__ t1,
                                                    const float* __restrict__ t1d,
                                                    const float* __restrict__ tf,
                                                    float* __restrict__ Tf32,
                                                    __bf16* __restrict__ Tbf, long n) {
  long i = (long)blockIdx.x * 256 + threadIdx.x;
  if (i >= n) return;
  float a = sigf(tf[i]);
  float v = t1[i] * a + t1d[i] * (1.f - a);
  Tf32[i] = v; Tbf[i] = (__bf16)v;
}

__global__ __launch_bounds__(256) void relu_bf_kernel(const float* __restrict__ x,
                                                      __bf16* __restrict__ o, long n) {
  long i = (long)blockIdx.x * 256 + threadIdx.x;
  if (i >= n) return;
  float v = x[i];
  o[i] = (__bf16)(v > 0.f ? v : 0.f);
}

__global__ __launch_bounds__(256) void gate_kernel(const float* __restrict__ g2,
                                                   float* __restrict__ Sf,
                                                   __bf16* __restrict__ Sb, long n) {
  long i = (long)blockIdx.x * 256 + threadIdx.x;
  if (i >= n) return;
  float s = Sf[i];
  s = s + s * sigf(g2[i]);
  Sf[i] = s; Sb[i] = (__bf16)s;
}

__global__ __launch_bounds__(256) void tsblend_kernel(const float* __restrict__ Tf32,
                                                      const float* __restrict__ Sf,
                                                      const float* __restrict__ TSf,
                                                      __bf16* __restrict__ ob, long n) {
  long i = (long)blockIdx.x * 256 + threadIdx.x;
  if (i >= n) return;
  float a = sigf(TSf[i]);
  ob[i] = (__bf16)(Tf32[i] * a + Sf[i] * (1.f - a));
}

// ---------------------------------------------------------------------------
// Pooling conv (2x2xkz, stride = kernel) and inverse (transposed) conv.
// ---------------------------------------------------------------------------
__global__ __launch_bounds__(256) void pool_kernel(const __bf16* __restrict__ in,
                                                   const __bf16* __restrict__ w,
                                                   __bf16* __restrict__ out,
                                                   int C, int X, int Y, int Z, int T, int kz) {
  const int Xo = X / 2, Yo = Y / 2, Zo = Z / kz;
  const long No = (long)Xo * Yo * Zo * T;
  const long Ni = (long)X * Y * Z * T;
  long idx = (long)blockIdx.x * 256 + threadIdx.x;
  if (idx >= (long)C * No) return;
  int o = (int)(idx / No);
  long nv = idx % No;
  int t = (int)(nv % T); long q = nv / T;
  int zo = (int)(q % Zo); q /= Zo;
  int yo = (int)(q % Yo);
  int xo = (int)(q / Yo);
  float acc = 0.f;
  for (int ci = 0; ci < C; ++ci)
    for (int i = 0; i < 2; ++i)
      for (int j = 0; j < 2; ++j)
        for (int k = 0; k < kz; ++k) {
          long src = ((((long)(2 * xo + i) * Y + (2 * yo + j)) * Z + (kz * zo + k)) * T + t);
          acc += (float)w[(((size_t)o * C + ci) * 2 + i) * 2 * kz + (size_t)j * kz + k] *
                 (float)in[(size_t)ci * Ni + src];
        }
  out[idx] = (__bf16)acc;
}

__global__ __launch_bounds__(256) void up_kernel(const __bf16* __restrict__ in,
                                                 const __bf16* __restrict__ w,
                                                 __bf16* __restrict__ out,
                                                 int O, int C, int Xi, int Yi, int Zi,
                                                 int T, int kz) {
  const int Xo = 2 * Xi, Yo = 2 * Yi, Zo = kz * Zi;
  const long No = (long)Xo * Yo * Zo * T;
  const long Ni = (long)Xi * Yi * Zi * T;
  long idx = (long)blockIdx.x * 256 + threadIdx.x;
  if (idx >= (long)O * No) return;
  int o = (int)(idx / No);
  long nv = idx % No;
  int t = (int)(nv % T); long q = nv / T;
  int zo = (int)(q % Zo); q /= Zo;
  int yo = (int)(q % Yo);
  int xo = (int)(q / Yo);
  int i = xo & 1, xs = xo >> 1;
  int j = yo & 1, ys = yo >> 1;
  int k = (kz == 2) ? (zo & 1) : 0;
  int zs = (kz == 2) ? (zo >> 1) : zo;
  long src = (((long)xs * Yi + ys) * Zi + zs) * T + t;
  float acc = 0.f;
  for (int c = 0; c < C; ++c)
    acc += (float)w[(((size_t)o * C + c) * 2 + i) * 2 * kz + (size_t)j * kz + k] *
           (float)in[(size_t)c * Ni + src];
  out[idx] = (__bf16)acc;
}

// ---------------------------------------------------------------------------
// Host orchestration
// ---------------------------------------------------------------------------
namespace {

struct Arena { char* p; size_t off; };
inline void* aalloc(Arena& a, size_t bytes) {
  size_t al = (bytes + 255) & ~(size_t)255;
  void* r = a.p + a.off;
  a.off += al;
  return r;
}
inline long cdivl(long a, long b) { return (a + b - 1) / b; }

struct BW { const __bf16* w[9]; };  // ws wt wtd wtf wg1 wg2 wts wo [wp]

void gemm(hipStream_t s, const __bf16* W, const __bf16* act, float* out, int M, long N, int K) {
  dim3 g((unsigned)cdivl(N, 64), (unsigned)(M / 16));
  wmma_gemm_kernel<<<g, 128, 0, s>>>(W, act, out, M, (int)N, K);
}
void conv(hipStream_t s, const __bf16* W, const __bf16* act, float* out, int M, int K,
          int X, int Y, int Z, int T, int ntaps, int mode, int dil) {
  long N = (long)X * Y * Z * T;
  dim3 g((unsigned)cdivl(N, 32), (unsigned)(M / 16));
  wmma_conv_kernel<<<g, 32, 0, s>>>(W, act, out, M, K, X, Y, Z, T, ntaps, mode, dil);
}
void bn(hipStream_t s, float* x, float* mv, int C, long N, float* of, __bf16* ob) {
  zero_kernel<<<(unsigned)cdivl(2 * C, 256), 256, 0, s>>>(mv, 2 * C);
  long nb = cdivl(N, 256); if (nb > 256) nb = 256;
  dim3 g((unsigned)nb, (unsigned)C);
  bn_stats_kernel<<<g, 256, 0, s>>>(x, mv, C, N);
  bn_act_kernel<<<(unsigned)cdivl((long)C * N, 256), 256, 0, s>>>(x, mv, of, ob, C, N);
}

void run_block(hipStream_t s, Arena scr, const BW& p, const __bf16* xin,
               int Cin, int Mid, int Cout, int X, int Y, int Z, int T,
               __bf16* out_bf, float* out_f32, __bf16* pooled_bf, int kz) {
  const long N = (long)X * Y * Z * T;
  float*  mv   = (float*)aalloc(scr, sizeof(float) * 1024);
  float*  Sf   = (float*)aalloc(scr, sizeof(float) * (size_t)Mid * N);
  float*  T1f  = (float*)aalloc(scr, sizeof(float) * (size_t)Mid * N);
  float*  T1df = (float*)aalloc(scr, sizeof(float) * (size_t)Mid * N);
  float*  Tff  = (float*)aalloc(scr, sizeof(float) * (size_t)Mid * N);
  float*  Tblf = (float*)aalloc(scr, sizeof(float) * (size_t)Mid * N);
  float*  TSf  = (float*)aalloc(scr, sizeof(float) * (size_t)Mid * N);
  float*  Big  = (float*)aalloc(scr, sizeof(float) * (size_t)2 * Mid * N);  // g1 / out raw
  float*  G2f  = (float*)aalloc(scr, sizeof(float) * (size_t)Mid * N);
  __bf16* cat2 = (__bf16*)aalloc(scr, sizeof(__bf16) * (size_t)2 * Mid * N); // [T1|T1d]
  __bf16* cat4 = (__bf16*)aalloc(scr, sizeof(__bf16) * (size_t)2 * Mid * N); // [S|T]
  __bf16* g1b  = (__bf16*)aalloc(scr, sizeof(__bf16) * (size_t)2 * Mid * N);
  __bf16* cat5 = (__bf16*)aalloc(scr, sizeof(__bf16) * (size_t)(Mid + Cin) * N); // [TS|res]
  const long nMid = (long)Mid * N;

  // S = lrelu(bn(conv3x3x3(x)))
  conv(s, p.w[0], xin, Sf, Mid, Cin, X, Y, Z, T, 27, 0, 1);
  bn(s, Sf, mv, Mid, N, Sf, cat4);
  // T1 = lrelu(bn(conv1x1x1x3(x))) ; T1d = dilated(2)
  conv(s, p.w[1], xin, T1f, Mid, Cin, X, Y, Z, T, 3, 1, 1);
  bn(s, T1f, mv, Mid, N, T1f, cat2);
  conv(s, p.w[2], xin, T1df, Mid, Cin, X, Y, Z, T, 3, 1, 2);
  bn(s, T1df, mv, Mid, N, T1df, cat2 + (size_t)Mid * N);
  // Tf = lrelu(bn(W_tf @ [T1|T1d])); T = T1*sig(Tf) + T1d*(1-sig)
  gemm(s, p.w[3], cat2, Tff, Mid, N, 2 * Mid);
  bn(s, Tff, mv, Mid, N, Tff, nullptr);
  blend_kernel<<<(unsigned)cdivl(nMid, 256), 256, 0, s>>>(T1f, T1df, Tff, Tblf,
                                                          cat4 + (size_t)Mid * N, nMid);
  // gate = sig(W_g2 @ relu(W_g1 @ T)); S += S*gate
  gemm(s, p.w[4], cat4 + (size_t)Mid * N, Big, 2 * Mid, N, Mid);
  relu_bf_kernel<<<(unsigned)cdivl(2 * nMid, 256), 256, 0, s>>>(Big, g1b, 2 * nMid);
  gemm(s, p.w[5], g1b, G2f, Mid, N, 2 * Mid);
  gate_kernel<<<(unsigned)cdivl(nMid, 256), 256, 0, s>>>(G2f, Sf, cat4, nMid);
  // TS = lrelu(bn(W_ts @ [S|T])); A = sig(TS); TS = T*A + S*(1-A)
  gemm(s, p.w[6], cat4, TSf, Mid, N, 2 * Mid);
  bn(s, TSf, mv, Mid, N, TSf, nullptr);
  tsblend_kernel<<<(unsigned)cdivl(nMid, 256), 256, 0, s>>>(Tblf, Sf, TSf, cat5, nMid);
  hipMemcpyAsync(cat5 + (size_t)Mid * N, xin, sizeof(__bf16) * (size_t)Cin * N,
                 hipMemcpyDeviceToDevice, s);
  // out = lrelu(bn(W_o @ [TS|res]))
  gemm(s, p.w[7], cat5, Big, Cout, N, Mid + Cin);
  bn(s, Big, mv, Cout, N, out_f32, out_bf);
  if (pooled_bf) {
    long No = ((long)X / 2) * (Y / 2) * (Z / kz) * T * Cout;
    pool_kernel<<<(unsigned)cdivl(No, 256), 256, 0, s>>>(out_bf, p.w[8], pooled_bf,
                                                         Cout, X, Y, Z, T, kz);
  }
}

}  // namespace

extern "C" void kernel_launch(void* const* d_in, const int* in_sizes, int n_in,
                              void* d_out, int out_size, void* d_ws, size_t ws_size,
                              hipStream_t stream) {
  if (n_in < 121) return;
  (void)out_size; (void)ws_size;
  Arena ar{(char*)d_ws, 0};

  int ix = 0;
  // plain f32 -> bf16 conversion (x, pooling weights, up weights)
  auto cv = [&]() -> const __bf16* {
    long n = in_sizes[ix];
    __bf16* p = (__bf16*)aalloc(ar, sizeof(__bf16) * (size_t)n);
    cvt_bf16_kernel<<<(unsigned)cdivl(n, 256), 256, 0, stream>>>(
        (const float*)d_in[ix], p, n);
    ++ix;
    return p;
  };
  // repack [M][K][taps] f32 -> [taps][M][Kp] bf16 (Kp = roundup32, zero pad)
  auto rp = [&](int M, int K, int taps) -> const __bf16* {
    int Kp = (K + 31) & ~31;
    long tot = (long)taps * M * Kp;
    __bf16* p = (__bf16*)aalloc(ar, sizeof(__bf16) * (size_t)tot);
    repack_weight_kernel<<<(unsigned)cdivl(tot, 256), 256, 0, stream>>>(
        (const float*)d_in[ix], p, M, K, Kp, taps);
    ++ix;
    return p;
  };
  auto prep = [&](int Cin, int Mid, int Cout, bool pool) -> BW {
    BW b{};
    b.w[0] = rp(Mid, Cin, 27);
    b.w[1] = rp(Mid, Cin, 3);
    b.w[2] = rp(Mid, Cin, 3);
    b.w[3] = rp(Mid, 2 * Mid, 1);
    b.w[4] = rp(2 * Mid, Mid, 1);
    b.w[5] = rp(Mid, 2 * Mid, 1);
    b.w[6] = rp(Mid, 2 * Mid, 1);
    b.w[7] = rp(Cout, Cin + Mid, 1);
    if (pool) b.w[8] = cv();
    return b;
  };

  const __bf16* xbf = cv();  // input x (d_in[0])

  BW b111 = prep(16, 16, 16, false),   b112 = prep(16, 16, 32, true);
  BW b211 = prep(32, 32, 32, false),   b212 = prep(32, 32, 64, true);
  BW b311 = prep(64, 64, 64, false),   b312 = prep(64, 64, 64, true);
  BW b411 = prep(64, 64, 64, false),   b412 = prep(64, 64, 64, true);
  BW b511 = prep(64, 64, 64, false),   b512 = prep(64, 64, 64, false);
  const __bf16* up5w = cv();
  BW b421 = prep(128, 128, 64, false);
  const __bf16* up4w = cv();
  BW b321 = prep(128, 128, 64, false);
  const __bf16* up3w = cv();
  BW b221 = prep(128, 64, 64, false);
  const __bf16* up2w = cv();
  BW b121 = prep(64, 32, 16, false);

  // Dims per level: [X,Y,Z,T]
  const int T_ = 3;
  const long N1 = 64L * 64 * 16 * T_, N2 = 32L * 32 * 8 * T_, N3 = 16L * 16 * 4 * T_;
  const long N4 = 8L * 8 * 2 * T_,    N5 = 4L * 4 * 2 * T_;

  auto balloc = [&](long elems) { return (__bf16*)aalloc(ar, sizeof(__bf16) * (size_t)elems); };

  __bf16* s1    = balloc(16 * N1);
  __bf16* skip1 = balloc(32 * N1);
  __bf16* d2    = balloc(32 * N2);
  __bf16* d2b   = balloc(32 * N2);
  __bf16* skip2 = balloc(64 * N2);
  __bf16* d3    = balloc(64 * N3);
  __bf16* d3b   = balloc(64 * N3);
  __bf16* skip3 = balloc(64 * N3);
  __bf16* d4    = balloc(64 * N4);
  __bf16* d4b   = balloc(64 * N4);
  __bf16* skip4 = balloc(64 * N4);
  __bf16* d5    = balloc(64 * N5);
  __bf16* d5b   = balloc(64 * N5);
  __bf16* d5c   = balloc(64 * N5);
  __bf16* catU4 = balloc(128 * N4);
  __bf16* u4b   = balloc(64 * N4);
  __bf16* catU3 = balloc(128 * N3);
  __bf16* u3b   = balloc(64 * N3);
  __bf16* catU2 = balloc(128 * N2);
  __bf16* u2b   = balloc(64 * N2);
  __bf16* catU1 = balloc(64 * N1);
  __bf16* fout  = balloc(16 * N1);

  // ---- Encoder ----
  run_block(stream, ar, b111, xbf, 16, 16, 16, 64, 64, 16, T_, s1,    nullptr, nullptr, 0);
  run_block(stream, ar, b112, s1,  16, 16, 32, 64, 64, 16, T_, skip1, nullptr, d2,      2);
  run_block(stream, ar, b211, d2,  32, 32, 32, 32, 32, 8,  T_, d2b,   nullptr, nullptr, 0);
  run_block(stream, ar, b212, d2b, 32, 32, 64, 32, 32, 8,  T_, skip2, nullptr, d3,      2);
  run_block(stream, ar, b311, d3,  64, 64, 64, 16, 16, 4,  T_, d3b,   nullptr, nullptr, 0);
  run_block(stream, ar, b312, d3b, 64, 64, 64, 16, 16, 4,  T_, skip3, nullptr, d4,      2);
  run_block(stream, ar, b411, d4,  64, 64, 64, 8,  8,  2,  T_, d4b,   nullptr, nullptr, 0);
  run_block(stream, ar, b412, d4b, 64, 64, 64, 8,  8,  2,  T_, skip4, nullptr, d5,      1);
  run_block(stream, ar, b511, d5,  64, 64, 64, 4,  4,  2,  T_, d5b,   nullptr, nullptr, 0);
  run_block(stream, ar, b512, d5b, 64, 64, 64, 4,  4,  2,  T_, d5c,   nullptr, nullptr, 0);

  // ---- Decoder ----
  up_kernel<<<(unsigned)cdivl(64 * N4, 256), 256, 0, stream>>>(d5c, up5w, catU4,
                                                               64, 64, 4, 4, 2, T_, 1);
  hipMemcpyAsync(catU4 + (size_t)64 * N4, skip4, sizeof(__bf16) * (size_t)64 * N4,
                 hipMemcpyDeviceToDevice, stream);
  run_block(stream, ar, b421, catU4, 128, 128, 64, 8, 8, 2, T_, u4b, nullptr, nullptr, 0);

  up_kernel<<<(unsigned)cdivl(64 * N3, 256), 256, 0, stream>>>(u4b, up4w, catU3,
                                                               64, 64, 8, 8, 2, T_, 2);
  hipMemcpyAsync(catU3 + (size_t)64 * N3, skip3, sizeof(__bf16) * (size_t)64 * N3,
                 hipMemcpyDeviceToDevice, stream);
  run_block(stream, ar, b321, catU3, 128, 128, 64, 16, 16, 4, T_, u3b, nullptr, nullptr, 0);

  up_kernel<<<(unsigned)cdivl(64 * N2, 256), 256, 0, stream>>>(u3b, up3w, catU2,
                                                               64, 64, 16, 16, 4, T_, 2);
  hipMemcpyAsync(catU2 + (size_t)64 * N2, skip2, sizeof(__bf16) * (size_t)64 * N2,
                 hipMemcpyDeviceToDevice, stream);
  run_block(stream, ar, b221, catU2, 128, 64, 64, 32, 32, 8, T_, u2b, nullptr, nullptr, 0);

  up_kernel<<<(unsigned)cdivl(32 * N1, 256), 256, 0, stream>>>(u2b, up2w, catU1,
                                                               32, 64, 32, 32, 8, T_, 2);
  hipMemcpyAsync(catU1 + (size_t)32 * N1, skip1, sizeof(__bf16) * (size_t)32 * N1,
                 hipMemcpyDeviceToDevice, stream);
  run_block(stream, ar, b121, catU1, 64, 32, 16, 64, 64, 16, T_, fout, (float*)d_out,
            nullptr, 0);
}
